// HGT_37709812859008
// MI455X (gfx1250) — matrix operand order repaired and checked by reading, not compiled, not run on hardware
//
#include <hip/hip_runtime.h>
#include <math.h>

// ---------------------------------------------------------------------------
// HGT forward for MI455X (gfx1250).  Dense projections run on the WMMA pipe
// (v_wmma_f32_16x16x32_bf16, split-bf16 for fp32-class accuracy).  The f32->
// bf16 hi/lo split is done ONCE at global->LDS staging time, stored as packed
// bf16x2 pairs along K, so WMMA fragments are built with raw ds_load_b32 and
// bit-casts (no conversion VALU on the critical path).  Edge softmax /
// aggregation is the memory-bound part and uses coalesced 32-float row
// segments with L2 atomics.
// ---------------------------------------------------------------------------

typedef __attribute__((ext_vector_type(16))) __bf16    v16bf;
typedef __attribute__((ext_vector_type(2)))  __bf16    bf16x2;
typedef __attribute__((ext_vector_type(8)))  float     v8f;
typedef __attribute__((ext_vector_type(8)))  unsigned  v8u;

#define NWEB   100000
#define NUSR   200000
#define C_DIM  128
#define H_DIM  4
#define D_DIM  32

// split v into bf16 hi + residual lo for a K-consecutive pair, packed as dwords
__device__ __forceinline__ unsigned pack_hilo(float v0, float v1, unsigned* lo_out) {
    __bf16 h0 = (__bf16)v0, h1 = (__bf16)v1;
    __bf16 l0 = (__bf16)(v0 - (float)h0), l1 = (__bf16)(v1 - (float)h1);
    bf16x2 hp = { h0, h1 }, lp = { l0, l1 };
    *lo_out = __builtin_bit_cast(unsigned, lp);
    return __builtin_bit_cast(unsigned, hp);
}

// ======================= WMMA GEMM: Out[M,128] = act(A[M,K] @ W[K,128] + b)
// Block: 256 threads = 8 waves.  Block tile 64 rows x 128 cols.
// Wave (w&3) picks a 16-row strip, (w>>2) picks a 64-col half; 4 col tiles.
// LDS holds pre-split packed-bf16 K-pairs:
//   Ah/Al[row][kpair]      (64 x 16, stride 17)
//   Wh/Wl[kpair][col]      (16 x 128, stride 130)
// Fragment VGPR j of the 16x16x32 bf16 fragments is exactly one K-pair dword:
//   A: kpair(j) = (j<4 ? j : j+4) + half*4        (ISA A 16x32 layout)
//   B: kpair(j) = (j<4 ? j : j+4) + half*4        (same K order as A)
__global__ __launch_bounds__(256)
void gemm_bias_act(const float* __restrict__ A, const float* __restrict__ W,
                   const float* __restrict__ bias, float* __restrict__ Out,
                   int M, int K, int act) {
    __shared__ unsigned Ah[64 * 17], Al[64 * 17];
    __shared__ unsigned Wh[16 * 130], Wl[16 * 130];

    const int tid      = threadIdx.x;
    const int lane     = tid & 31;
    const int wave     = tid >> 5;
    const int blockRow = blockIdx.x * 64;
    const int waveRow  = (wave & 3) * 16;
    const int waveCol  = (wave >> 2) * 64;
    const int r        = lane & 15;   // A row / B col within tile
    const int half     = lane >> 4;   // lane half selects K sub-block

    v8f acc[4] = {};

    for (int k0 = 0; k0 < K; k0 += 32) {
        // --- stage A tile (64 rows x 16 K-pairs), split+pack once ---
        for (int p = tid; p < 64 * 16; p += 256) {
            int rr = p >> 4, kp = p & 15;
            int grow = blockRow + rr;
            float v0 = 0.0f, v1 = 0.0f;
            if (grow < M) {
                const float* ap = A + (size_t)grow * K + k0 + 2 * kp;
                v0 = ap[0]; v1 = ap[1];
            }
            unsigned lo, hi = pack_hilo(v0, v1, &lo);
            Ah[rr * 17 + kp] = hi;
            Al[rr * 17 + kp] = lo;
        }
        // --- stage W tile (16 K-pairs x 128 cols), split+pack once ---
        for (int p = tid; p < 16 * 128; p += 256) {
            int kp = p >> 7, cc = p & 127;
            const float* wp = W + (size_t)(k0 + 2 * kp) * 128 + cc;
            unsigned lo, hi = pack_hilo(wp[0], wp[128], &lo);
            Wh[kp * 130 + cc] = hi;
            Wl[kp * 130 + cc] = lo;
        }
        // prefetch next W tile into L2 while we compute (global_prefetch_b8)
        if (k0 + 32 < K)
            __builtin_prefetch(W + (size_t)(k0 + 32 + (tid >> 7)) * 128 + (tid & 127), 0, 1);
        __syncthreads();

        // A fragments: 8 ds_load_b32 each, bit-cast to v16bf
        v8u ahu, alu;
        const int arow = (waveRow + r) * 17;
#pragma unroll
        for (int j = 0; j < 8; ++j) {
            int kp = ((j < 4) ? j : (j + 4)) + half * 4;
            ahu[j] = Ah[arow + kp];
            alu[j] = Al[arow + kp];
        }
        const v16bf a_hi = __builtin_bit_cast(v16bf, ahu);
        const v16bf a_lo = __builtin_bit_cast(v16bf, alu);

#pragma unroll
        for (int t = 0; t < 4; ++t) {
            const int col = waveCol + t * 16 + r;
            v8u bhu, blu;
#pragma unroll
            for (int j = 0; j < 8; ++j) {
                bhu[j] = Wh[(((j < 4) ? j : (j + 4)) + half * 4) * 130 + col];
                blu[j] = Wl[(((j < 4) ? j : (j + 4)) + half * 4) * 130 + col];
            }
            const v16bf b_hi = __builtin_bit_cast(v16bf, bhu);
            const v16bf b_lo = __builtin_bit_cast(v16bf, blu);
            acc[t] = __builtin_amdgcn_wmma_f32_16x16x32_bf16(
                false, a_hi, false, b_hi, (short)0, acc[t], false, false);
            acc[t] = __builtin_amdgcn_wmma_f32_16x16x32_bf16(
                false, a_hi, false, b_lo, (short)0, acc[t], false, false);
            acc[t] = __builtin_amdgcn_wmma_f32_16x16x32_bf16(
                false, a_lo, false, b_hi, (short)0, acc[t], false, false);
        }
        __syncthreads();
    }

    // Epilogue. D layout: lane&15 = col, row = 8*(lane>>4) + vgpr index.
#pragma unroll
    for (int t = 0; t < 4; ++t) {
        const int col = waveCol + t * 16 + r;
        const float bv = bias[col];
#pragma unroll
        for (int j = 0; j < 8; ++j) {
            int row = blockRow + waveRow + half * 8 + j;
            if (row < M) {
                float v = acc[t][j] + bv;
                if (act == 1) v = fmaxf(v, 0.0f);
                Out[(size_t)row * 128 + col] = v;
            }
        }
    }
}

// ======================= per-head relation transform: out[n,h,:]=in[n,h,:]@R[h]
__global__ void rel_transform(const float* __restrict__ in, const float* __restrict__ R,
                              float* __restrict__ out, int n_total) {
    int tid = blockIdx.x * blockDim.x + threadIdx.x;   // over N*128
    if (tid >= n_total) return;
    int e = tid & 31;
    int h = (tid >> 5) & 3;
    int n = tid >> 7;
    const float* Rh  = R + h * D_DIM * D_DIM;
    const float* row = in + (size_t)n * C_DIM + h * D_DIM;
    float s = 0.0f;
#pragma unroll
    for (int d = 0; d < 32; ++d) s += row[d] * Rh[d * 32 + e];
    out[tid] = s;
}

// ======================= ordered-int float max key
__device__ __forceinline__ unsigned fkey(float f) {
    unsigned u = __float_as_uint(f);
    return (u & 0x80000000u) ? ~u : (u | 0x80000000u);
}
__device__ __forceinline__ float funkey(unsigned k) {
    unsigned u = (k & 0x80000000u) ? (k & 0x7FFFFFFFu) : ~k;
    return __uint_as_float(u);
}

// ======================= edge pass 1: alpha + segment max
__global__ void edge_alpha(const int* __restrict__ ei, int E,
                           const float* __restrict__ q, const float* __restrict__ krel,
                           const float* __restrict__ p, float* __restrict__ alpha,
                           unsigned* __restrict__ m) {
    int tid = blockIdx.x * blockDim.x + threadIdx.x;   // over E*H
    if (tid >= E * H_DIM) return;
    int h = tid & 3, e = tid >> 2;
    int si = ei[e], di = ei[E + e];
    const float* qr = q    + (size_t)di * C_DIM + h * D_DIM;
    const float* kr = krel + (size_t)si * C_DIM + h * D_DIM;
    float s = 0.0f;
#pragma unroll
    for (int d = 0; d < 32; ++d) s += qr[d] * kr[d];
    float a = s * p[h] * 0.17677669529663687f;   // 1/sqrt(D)
    alpha[tid] = a;
    atomicMax(&m[(size_t)di * H_DIM + h], fkey(a));
}

// ======================= edge pass 2: exp + segment sum
__global__ void edge_exp(const int* __restrict__ ei, int E,
                         float* __restrict__ alpha, const unsigned* __restrict__ m,
                         float* __restrict__ ssum) {
    int tid = blockIdx.x * blockDim.x + threadIdx.x;
    if (tid >= E * H_DIM) return;
    int h = tid & 3, e = tid >> 2;
    int di = ei[E + e];
    float mv = funkey(m[(size_t)di * H_DIM + h]);
    float ew = __expf(alpha[tid] - mv);
    alpha[tid] = ew;
    atomicAdd(&ssum[(size_t)di * H_DIM + h], ew);
}

// ======================= edge pass 3: weighted scatter of V
__global__ void edge_scatter(const int* __restrict__ ei, int E,
                             const float* __restrict__ ew, const float* __restrict__ ssum,
                             const float* __restrict__ vrel, float* __restrict__ agg) {
    int tid = blockIdx.x * blockDim.x + threadIdx.x;
    if (tid >= E * H_DIM) return;
    int h = tid & 3, e = tid >> 2;
    int si = ei[e], di = ei[E + e];
    float w = ew[tid] / ssum[(size_t)di * H_DIM + h];
    const float* vr = vrel + (size_t)si * C_DIM + h * D_DIM;
    float*       ar = agg  + (size_t)di * C_DIM + h * D_DIM;
#pragma unroll
    for (int d = 0; d < 32; ++d) atomicAdd(&ar[d], vr[d] * w);
}

// ======================= exact GELU, in place
__global__ void gelu_ip(float* __restrict__ x, int n) {
    int tid = blockIdx.x * blockDim.x + threadIdx.x;
    if (tid >= n) return;
    float v = x[tid];
    x[tid] = 0.5f * v * (1.0f + erff(v * 0.70710678118654752f));
}

// ======================= skip blend: x = sk*aout + (1-sk)*x
__global__ void skip_blend(const float* __restrict__ skip, const float* __restrict__ aout,
                           float* __restrict__ x, int n) {
    int tid = blockIdx.x * blockDim.x + threadIdx.x;
    if (tid >= n) return;
    float sk = 1.0f / (1.0f + __expf(-skip[0]));
    x[tid] = sk * aout[tid] + (1.0f - sk) * x[tid];
}

// ======================= 128->2 head + log_softmax
__global__ void out_logsoftmax(const float* __restrict__ x, const float* __restrict__ w,
                               const float* __restrict__ b, float* __restrict__ out, int n) {
    int tid = blockIdx.x * blockDim.x + threadIdx.x;
    if (tid >= n) return;
    const float* row = x + (size_t)tid * C_DIM;
    float z0 = b[0], z1 = b[1];
#pragma unroll 16
    for (int i = 0; i < 128; ++i) {
        float v = row[i];
        z0 += v * w[i * 2 + 0];
        z1 += v * w[i * 2 + 1];
    }
    float mm  = fmaxf(z0, z1);
    float lse = mm + logf(__expf(z0 - mm) + __expf(z1 - mm));
    out[tid * 2 + 0] = z0 - lse;
    out[tid * 2 + 1] = z1 - lse;
}

// ===========================================================================
static inline int g1(int n, int b) { return (n + b - 1) / b; }

extern "C" void kernel_launch(void* const* d_in, const int* in_sizes, int n_in,
                              void* d_out, int out_size, void* d_ws, size_t ws_size,
                              hipStream_t stream) {
    (void)n_in; (void)out_size; (void)ws_size;

    // ---- inputs: dict order ----
    const float* x_web_in = (const float*)d_in[0];   // 100000 x 256
    const float* x_usr_in = (const float*)d_in[1];   // 200000 x 128
    const int*   ei_fol   = (const int*)d_in[2];
    const int*   ei_men   = (const int*)d_in[3];
    const int*   ei_lnk   = (const int*)d_in[4];
    const int    E        = in_sizes[2] / 2;

    // ---- params: JAX pytree flatten of nested dicts (sorted keys).
    // Top: layers(list) < lin_in < lin_out.
    // Layer: a, k, q, rel(follows<links<mentions; a_rel<m_rel<p), skip, v.
    // node-type dicts: users < websites; linears: b < w.  27 leaves/layer.
    const int P = 5;
    auto prm = [&](int i) { return (const float*)d_in[P + i]; };
    const int L0 = 0, L1 = 27;
    const int LIN_IN = 54;   // users.b, users.w, websites.b, websites.w
    const int LIN_OUT = 58;  // b, w

    // ---- workspace bump allocator (floats) ----
    float* ws = (float*)d_ws;
    size_t off = 0;
    auto alloc = [&](size_t n) { float* p = ws + off; off += (n + 63) & ~(size_t)63; return p; };
    const size_t XW = (size_t)NWEB * C_DIM, XU = (size_t)NUSR * C_DIM;

    float* xw   = alloc(XW);  float* xu   = alloc(XU);
    float* kw   = alloc(XW);  float* qw   = alloc(XW);  float* vw = alloc(XW);
    float* ku   = alloc(XU);  float* qu   = alloc(XU);  float* vu = alloc(XU);
    float* krel = alloc(XU);  float* vrel = alloc(XU);          // sized for max src (users)
    float* aggw = alloc(XW);  float* aggu = alloc(XU);
    float* ewb  = alloc((size_t)E * H_DIM);
    unsigned* mbuf = (unsigned*)alloc((size_t)NUSR * H_DIM);
    float*    sbuf = alloc((size_t)NUSR * H_DIM);

    const dim3 B(256);

    // ---- input projections + relu ----
    gemm_bias_act<<<dim3(g1(NWEB, 64)), B, 0, stream>>>(
        x_web_in, prm(LIN_IN + 3), prm(LIN_IN + 2), xw, NWEB, 256, 1);
    gemm_bias_act<<<dim3(g1(NUSR, 64)), B, 0, stream>>>(
        x_usr_in, prm(LIN_IN + 1), prm(LIN_IN + 0), xu, NUSR, 128, 1);

    const int LB[2] = { L0, L1 };
    for (int l = 0; l < 2; ++l) {
        const int Lb = LB[l];
        // k/q/v projections (users then websites), offsets within layer:
        // a:0-3  k:4-7  q:8-11  rel:12-20  skip:21-22  v:23-26
        gemm_bias_act<<<dim3(g1(NUSR, 64)), B, 0, stream>>>(xu, prm(Lb + 5),  prm(Lb + 4),  ku, NUSR, 128, 0);
        gemm_bias_act<<<dim3(g1(NWEB, 64)), B, 0, stream>>>(xw, prm(Lb + 7),  prm(Lb + 6),  kw, NWEB, 128, 0);
        gemm_bias_act<<<dim3(g1(NUSR, 64)), B, 0, stream>>>(xu, prm(Lb + 9),  prm(Lb + 8),  qu, NUSR, 128, 0);
        gemm_bias_act<<<dim3(g1(NWEB, 64)), B, 0, stream>>>(xw, prm(Lb + 11), prm(Lb + 10), qw, NWEB, 128, 0);
        gemm_bias_act<<<dim3(g1(NUSR, 64)), B, 0, stream>>>(xu, prm(Lb + 24), prm(Lb + 23), vu, NUSR, 128, 0);
        gemm_bias_act<<<dim3(g1(NWEB, 64)), B, 0, stream>>>(xw, prm(Lb + 26), prm(Lb + 25), vw, NWEB, 128, 0);

        hipMemsetAsync(aggw, 0, XW * sizeof(float), stream);
        hipMemsetAsync(aggu, 0, XU * sizeof(float), stream);

        // relation table: {ei, k_src, v_src, Nsrc, q_dst, agg_dst, Ndst, a_rel, m_rel, p}
        struct Rel { const int* ei; const float *ks, *vs; int ns;
                     const float* qd; float* ag; int nd; int ar, mr, pp; };
        const Rel rels[3] = {
            { ei_fol, ku, vu, NUSR, qu, aggu, NUSR, Lb + 12, Lb + 13, Lb + 14 }, // follows
            { ei_men, ku, vu, NUSR, qw, aggw, NWEB, Lb + 18, Lb + 19, Lb + 20 }, // mentions
            { ei_lnk, kw, vw, NWEB, qw, aggw, NWEB, Lb + 15, Lb + 16, Lb + 17 }, // links
        };
        for (int rIdx = 0; rIdx < 3; ++rIdx) {
            const Rel& R = rels[rIdx];
            rel_transform<<<dim3(g1(R.ns * C_DIM, 256)), B, 0, stream>>>(R.ks, prm(R.ar), krel, R.ns * C_DIM);
            rel_transform<<<dim3(g1(R.ns * C_DIM, 256)), B, 0, stream>>>(R.vs, prm(R.mr), vrel, R.ns * C_DIM);
            hipMemsetAsync(mbuf, 0, (size_t)R.nd * H_DIM * sizeof(unsigned), stream); // key 0 < any finite
            hipMemsetAsync(sbuf, 0, (size_t)R.nd * H_DIM * sizeof(float), stream);
            const int eg = g1(E * H_DIM, 256);
            edge_alpha  <<<dim3(eg), B, 0, stream>>>(R.ei, E, R.qd, krel, prm(R.pp), ewb, mbuf);
            edge_exp    <<<dim3(eg), B, 0, stream>>>(R.ei, E, ewb, mbuf, sbuf);
            edge_scatter<<<dim3(eg), B, 0, stream>>>(R.ei, E, ewb, sbuf, vrel, R.ag);
        }

        // epilogue: gelu(agg) -> a-linear -> skip blend (a_out reuses q buffers)
        gelu_ip<<<dim3(g1((int)XU, 256)), B, 0, stream>>>(aggu, (int)XU);
        gelu_ip<<<dim3(g1((int)XW, 256)), B, 0, stream>>>(aggw, (int)XW);
        gemm_bias_act<<<dim3(g1(NUSR, 64)), B, 0, stream>>>(aggu, prm(Lb + 1), prm(Lb + 0), qu, NUSR, 128, 0);
        gemm_bias_act<<<dim3(g1(NWEB, 64)), B, 0, stream>>>(aggw, prm(Lb + 3), prm(Lb + 2), qw, NWEB, 128, 0);
        skip_blend<<<dim3(g1((int)XU, 256)), B, 0, stream>>>(prm(Lb + 21), qu, xu, (int)XU);
        skip_blend<<<dim3(g1((int)XW, 256)), B, 0, stream>>>(prm(Lb + 22), qw, xw, (int)XW);
    }

    // ---- output head + log_softmax over websites ----
    out_logsoftmax<<<dim3(g1(NWEB, 256)), B, 0, stream>>>(
        xw, prm(LIN_OUT + 1), prm(LIN_OUT + 0), (float*)d_out, NWEB);
}
